// BasicDMPNN_326417514977
// MI455X (gfx1250) — matrix operand order repaired and checked by reading, not compiled, not run on hardware
//
#include <hip/hip_runtime.h>
#include <hip/hip_bf16.h>

typedef __attribute__((ext_vector_type(16))) __bf16 v16bf;
typedef __attribute__((ext_vector_type(8)))  float  v8f;
typedef __attribute__((ext_vector_type(4)))  __bf16 v4bf;

#define ATOM_EMB 64
#define BOND_EMB 16
#define MSG_DIM  64
#define KPAD     160   // 64 atom + 16 bond + 64 agg + 16 zero-pad
#define OUT_HIDDEN 128

// ---------------------------------------------------------------------------
// helpers
// ---------------------------------------------------------------------------
__device__ __forceinline__ v16bf load_a_frag(const __bf16* row, int kb, int h) {
    // 16-bit A 16x32 layout: lane(m,h): K = kb + h*8 + {0..7}  and  kb+16+h*8+{0..7}
    union { uint4 q[2]; v16bf v; } u;
    u.q[0] = *reinterpret_cast<const uint4*>(row + kb + h * 8);
    u.q[1] = *reinterpret_cast<const uint4*>(row + kb + 16 + h * 8);
    return u.v;
}

__device__ __forceinline__ v16bf load_b_frag(const __bf16* w, int n, int kb, int h) {
    // 16-bit B 32x16 layout: lane holds column n, K = kb + h*8 + {0..7}  and  kb+16+h*8+{0..7}
    const __bf16* p = w + n * KPAD + kb + h * 8;
    union { uint4 q[2]; v16bf v; } u;
    u.q[0] = *reinterpret_cast<const uint4*>(p);
    u.q[1] = *reinterpret_cast<const uint4*>(p + 16);
    return u.v;
}

// ---------------------------------------------------------------------------
// build padded, transposed bf16 weight panels: Wt[n][k], k in [0,160)
//   Wt0: k<80  -> Wi[k][n], else 0     (initial pass, agg part multiplied by 0)
//   WtU: k<144 -> Wu[k][n], else 0     (update passes)
// ---------------------------------------------------------------------------
__global__ void dmpnn_prep_weights(const float* __restrict__ Wi,
                                   const float* __restrict__ Wu,
                                   __bf16* __restrict__ Wt0,
                                   __bf16* __restrict__ WtU) {
    int t = blockIdx.x * blockDim.x + threadIdx.x;
    if (t >= MSG_DIM * KPAD) return;
    int n = t / KPAD, k = t % KPAD;
    Wt0[t] = (k < 80)  ? (__bf16)Wi[k * MSG_DIM + n] : (__bf16)0.f;
    WtU[t] = (k < 144) ? (__bf16)Wu[k * MSG_DIM + n] : (__bf16)0.f;
}

__global__ void dmpnn_zero_f4(float4* __restrict__ p, int n4) {
    int i = blockIdx.x * blockDim.x + threadIdx.x;
    if (i < n4) p[i] = make_float4(0.f, 0.f, 0.f, 0.f);
}

// ---------------------------------------------------------------------------
// fused edge message pass:
//   per edge e: row = [bf16(atom_table[x[src]]) | bf16(bond_table[ea]) |
//                      bf16(agg_in[src]) | 0]            (K = 160)
//   msg = relu(row @ Wt^T + bias)                         (WMMA bf16->f32)
//   agg_out[dst] += msg                                   (f32 atomics, L2)
// One wave owns one 16-edge x 64-out tile: 5 K-steps x 4 N-tiles = 20 WMMAs.
// ---------------------------------------------------------------------------
__global__ __launch_bounds__(256)
void dmpnn_edge_wmma(const int*   __restrict__ src,
                     const int*   __restrict__ dst,
                     const int*   __restrict__ x,
                     const int*   __restrict__ eattr,
                     const float* __restrict__ atom_table,
                     const float* __restrict__ bond_table,
                     const __bf16* __restrict__ Wt,
                     const float* __restrict__ bias,
                     const float* __restrict__ agg_in,
                     float*       __restrict__ agg_out,
                     int E) {
    __shared__ __bf16 Wlds[MSG_DIM * KPAD];   // 20 KB
    __shared__ __bf16 Alds[128 * KPAD];       // 40 KB (8 tiles of 16 edges)

    const int tid = threadIdx.x;

    // stage weights (20480 B = 1280 uint4)
    {
        const uint4* wg = reinterpret_cast<const uint4*>(Wt);
        uint4*       wl = reinterpret_cast<uint4*>(Wlds);
        for (int i = tid; i < 1280; i += 256) wl[i] = wg[i];
    }

    // build the 128 x 160 bf16 A panel: thread = (edge_local, half-row)
    const int  el = tid >> 1;          // 0..127
    const int  hb = tid & 1;           // 0: atom+bond, 1: agg+pad
    const long e  = (long)blockIdx.x * 128 + el;
    __bf16* base  = &Alds[el * KPAD];
    if (e < E) {
        const int s = src[e];
        if (hb == 0) {
            const int xi = x[s];
            const float4* a4 = reinterpret_cast<const float4*>(atom_table + xi * ATOM_EMB);
            #pragma unroll
            for (int i = 0; i < 16; ++i) {
                float4 f = a4[i];
                *reinterpret_cast<v4bf*>(base + 4 * i) =
                    (v4bf){(__bf16)f.x, (__bf16)f.y, (__bf16)f.z, (__bf16)f.w};
            }
            const int b = eattr[e];
            const float4* b4 = reinterpret_cast<const float4*>(bond_table + b * BOND_EMB);
            #pragma unroll
            for (int i = 0; i < 4; ++i) {
                float4 f = b4[i];
                *reinterpret_cast<v4bf*>(base + 64 + 4 * i) =
                    (v4bf){(__bf16)f.x, (__bf16)f.y, (__bf16)f.z, (__bf16)f.w};
            }
        } else {
            const float4* g4 = reinterpret_cast<const float4*>(agg_in + (long)s * MSG_DIM);
            #pragma unroll
            for (int i = 0; i < 16; ++i) {
                float4 f = g4[i];
                *reinterpret_cast<v4bf*>(base + 80 + 4 * i) =
                    (v4bf){(__bf16)f.x, (__bf16)f.y, (__bf16)f.z, (__bf16)f.w};
            }
            const v4bf z4 = (v4bf){(__bf16)0.f, (__bf16)0.f, (__bf16)0.f, (__bf16)0.f};
            #pragma unroll
            for (int i = 0; i < 4; ++i)
                *reinterpret_cast<v4bf*>(base + 144 + 4 * i) = z4;
        }
    } else {
        const v4bf z4 = (v4bf){(__bf16)0.f, (__bf16)0.f, (__bf16)0.f, (__bf16)0.f};
        #pragma unroll
        for (int i = 0; i < 20; ++i)
            *reinterpret_cast<v4bf*>(base + hb * 80 + 4 * i) = z4;
    }
    __syncthreads();

    // ---- WMMA: wave wv computes edges [wv*16, wv*16+16) x all 64 outputs ----
    const int lane = tid & 31;
    const int wv   = tid >> 5;
    const int m    = lane & 15;        // A row / D column index
    const int h    = lane >> 4;        // lane half

    const __bf16* Arow = &Alds[(wv * 16 + m) * KPAD];

    v8f acc[4] = {};                   // 4 N-tiles of 16 columns
    #pragma unroll
    for (int kk = 0; kk < 5; ++kk) {
        const int kb = kk * 32;
        v16bf a = load_a_frag(Arow, kb, h);
        #pragma unroll
        for (int nt = 0; nt < 4; ++nt) {
            v16bf b = load_b_frag(Wlds, nt * 16 + m, kb, h);
            acc[nt] = __builtin_amdgcn_wmma_f32_16x16x32_bf16(
                false, a, false, b, (short)0, acc[nt], false, false);
        }
    }

    float bsv[4];
    #pragma unroll
    for (int nt = 0; nt < 4; ++nt) bsv[nt] = bias[nt * 16 + m];

    // D layout: vgpr r, lane half h -> edge row (h*8 + r), column n = m (+nt*16)
    const long e0 = (long)blockIdx.x * 128 + wv * 16;
    #pragma unroll
    for (int r = 0; r < 8; ++r) {
        const long ee = e0 + h * 8 + r;
        if (ee < E) {
            const int d = dst[ee];
            float* orow = agg_out + (long)d * MSG_DIM;
            #pragma unroll
            for (int nt = 0; nt < 4; ++nt) {
                float v = acc[nt][r] + bsv[nt];
                v = v > 0.f ? v : 0.f;
                unsafeAtomicAdd(orow + nt * 16 + m, v);   // global_atomic_add_f32 (L2)
            }
        }
    }
}

// ---------------------------------------------------------------------------
// node_state -> mol_state segment sum (batch is sorted; atomics hit L2)
// ---------------------------------------------------------------------------
__global__ void dmpnn_mol_seg(const float* __restrict__ node_state,
                              const int*   __restrict__ batch,
                              float*       __restrict__ mol,
                              int total) {
    int idx = blockIdx.x * blockDim.x + threadIdx.x;
    if (idx >= total) return;
    int node = idx >> 6;
    int c    = idx & 63;
    int mm   = batch[node];
    unsafeAtomicAdd(&mol[(long)mm * MSG_DIM + c], node_state[idx]);
}

// ---------------------------------------------------------------------------
// readout: out[m] = relu(mol[m] @ W1 + b1) @ W2 + b2  (one block per molecule)
// ---------------------------------------------------------------------------
__global__ __launch_bounds__(OUT_HIDDEN)
void dmpnn_readout(const float* __restrict__ mol,
                   const float* __restrict__ W1, const float* __restrict__ b1,
                   const float* __restrict__ W2, const float* __restrict__ b2,
                   float* __restrict__ out) {
    __shared__ float red[OUT_HIDDEN];
    const int j = threadIdx.x;
    const int mm = blockIdx.x;
    const float* mr = mol + (long)mm * MSG_DIM;
    float a = b1[j];
    #pragma unroll
    for (int i = 0; i < MSG_DIM; ++i) a += mr[i] * W1[i * OUT_HIDDEN + j];
    a = a > 0.f ? a : 0.f;
    red[j] = a * W2[j];
    __syncthreads();
    for (int s = OUT_HIDDEN / 2; s > 0; s >>= 1) {
        if (j < s) red[j] += red[j + s];
        __syncthreads();
    }
    if (j == 0) out[mm] = red[0] + b2[0];
}

// ---------------------------------------------------------------------------
extern "C" void kernel_launch(void* const* d_in, const int* in_sizes, int n_in,
                              void* d_out, int out_size, void* d_ws, size_t ws_size,
                              hipStream_t stream) {
    const int*   x          = (const int*)d_in[0];
    const int*   edge_attr  = (const int*)d_in[1];
    const int*   edge_index = (const int*)d_in[2];
    const int*   batch      = (const int*)d_in[3];
    const float* atom_table = (const float*)d_in[4];
    const float* bond_table = (const float*)d_in[5];
    const float* Wi         = (const float*)d_in[6];
    const float* bi         = (const float*)d_in[7];
    const float* Wu         = (const float*)d_in[8];
    const float* bu         = (const float*)d_in[9];
    const float* W1         = (const float*)d_in[10];
    const float* b1         = (const float*)d_in[11];
    const float* W2         = (const float*)d_in[12];
    const float* b2         = (const float*)d_in[13];
    float* out              = (float*)d_out;

    const int N = in_sizes[0];          // 100000 nodes
    const int E = in_sizes[1];          // 1.6M edges
    const int M = out_size;             // 2048 molecules
    const int* src = edge_index;
    const int* dst = edge_index + E;

    // workspace layout (all offsets 256B-aligned)
    char* ws = (char*)d_ws;
    size_t off = 0;
    const size_t aggBytes = (size_t)N * MSG_DIM * sizeof(float);
    float*  aggA = (float*)(ws + off);  off += aggBytes;                // 25.6 MB
    float*  aggB = (float*)(ws + off);  off += aggBytes;                // 25.6 MB
    __bf16* Wt0  = (__bf16*)(ws + off); off += (size_t)MSG_DIM * KPAD * 2; // 20 KB
    __bf16* WtU  = (__bf16*)(ws + off); off += (size_t)MSG_DIM * KPAD * 2; // 20 KB
    float*  mol  = (float*)(ws + off);  off += (size_t)M * MSG_DIM * sizeof(float);
    (void)ws_size; (void)n_in;

    // padded transposed bf16 weights
    {
        int tot = MSG_DIM * KPAD;
        dmpnn_prep_weights<<<(tot + 255) / 256, 256, 0, stream>>>(Wi, Wu, Wt0, WtU);
    }

    const int aggN4 = N * MSG_DIM / 4;
    const int molN4 = M * MSG_DIM / 4;
    dmpnn_zero_f4<<<(aggN4 + 255) / 256, 256, 0, stream>>>((float4*)aggA, aggN4);
    dmpnn_zero_f4<<<(aggN4 + 255) / 256, 256, 0, stream>>>((float4*)aggB, aggN4);
    dmpnn_zero_f4<<<(molN4 + 255) / 256, 256, 0, stream>>>((float4*)mol,  molN4);

    const int edgeBlocks = (E + 127) / 128;

    // pass 0: msg0 = relu(ab@Wi+bi); aggA = segsum(msg0)   (agg_in = zeros)
    dmpnn_edge_wmma<<<edgeBlocks, 256, 0, stream>>>(
        src, dst, x, edge_attr, atom_table, bond_table, Wt0, bi, aggB, aggA, E);

    // update passes 1..3: ping-pong
    dmpnn_zero_f4<<<(aggN4 + 255) / 256, 256, 0, stream>>>((float4*)aggB, aggN4);
    dmpnn_edge_wmma<<<edgeBlocks, 256, 0, stream>>>(
        src, dst, x, edge_attr, atom_table, bond_table, WtU, bu, aggA, aggB, E);

    dmpnn_zero_f4<<<(aggN4 + 255) / 256, 256, 0, stream>>>((float4*)aggA, aggN4);
    dmpnn_edge_wmma<<<edgeBlocks, 256, 0, stream>>>(
        src, dst, x, edge_attr, atom_table, bond_table, WtU, bu, aggB, aggA, E);

    dmpnn_zero_f4<<<(aggN4 + 255) / 256, 256, 0, stream>>>((float4*)aggB, aggN4);
    dmpnn_edge_wmma<<<edgeBlocks, 256, 0, stream>>>(
        src, dst, x, edge_attr, atom_table, bond_table, WtU, bu, aggA, aggB, E);

    // pass 4 + final segment sum: aggA = node_state = segsum(msg4)
    dmpnn_zero_f4<<<(aggN4 + 255) / 256, 256, 0, stream>>>((float4*)aggA, aggN4);
    dmpnn_edge_wmma<<<edgeBlocks, 256, 0, stream>>>(
        src, dst, x, edge_attr, atom_table, bond_table, WtU, bu, aggB, aggA, E);

    // molecule pooling + readout MLP
    {
        int tot = N * MSG_DIM;
        dmpnn_mol_seg<<<(tot + 255) / 256, 256, 0, stream>>>(aggA, batch, mol, tot);
    }
    dmpnn_readout<<<M, OUT_HIDDEN, 0, stream>>>(mol, W1, b1, W2, b2, out);
}
